// StructuralAttentionLayer_58737972740364
// MI455X (gfx1250) — matrix operand order repaired and checked by reading, compiled, not run on hardware
//
#include <hip/hip_runtime.h>
#include <math.h>

#define NN 40000   // nodes
#define EE 640000  // edges
#define DD 128     // D_IN == H*C
#define HH 8       // heads
#define CC 16      // channels per head

typedef __attribute__((ext_vector_type(16))) __bf16 v16bf;
typedef __attribute__((ext_vector_type(8)))  float  v8f;

// ---- float <-> order-preserving int (for atomicMax on floats) ----
__device__ __forceinline__ int f2oi(float f) {
  int i = __float_as_int(f);
  return (i >= 0) ? i : (i ^ 0x7FFFFFFF);
}
__device__ __forceinline__ float oi2f(int i) {
  return __int_as_float((i >= 0) ? i : (i ^ 0x7FFFFFFF));
}

// ============================================================================
// Kernel 1: fused dual GEMM  x = feat @ Wlin,  res = feat @ Wres
// bf16x3 split accumulation on v_wmma_f32_16x16x32_bf16 (wave32).
// Block = 256 threads = 8 waves; block owns rows [16b, 16b+16); wave w owns
// columns [16w, 16w+16). K loop: 4 steps of 32.
// ============================================================================
__global__ __launch_bounds__(256) void k_gemm_dual(
    const float* __restrict__ feat, const float* __restrict__ Wlin,
    const float* __restrict__ Wres, float* __restrict__ x,
    float* __restrict__ res) {
  const int row0 = blockIdx.x * 16;
  const int wave = threadIdx.x >> 5;
  const int lane = threadIdx.x & 31;
  const int col0 = wave * 16;
  const int lrow = lane & 15;
  const int hi   = lane >> 4;  // 0 or 1 (lane half)

  v8f clin = {0.f,0.f,0.f,0.f,0.f,0.f,0.f,0.f};
  v8f cres = {0.f,0.f,0.f,0.f,0.f,0.f,0.f,0.f};

  const float* arow = feat + (size_t)(row0 + lrow) * DD;
  const int colB = col0 + lrow;

  for (int kk = 0; kk < DD; kk += 32) {
    // ---- A tile (16x32): lane half selects K 0..15 / 16..31
    const int kA = kk + hi * 16;
    v16bf ah, al;
#pragma unroll
    for (int t = 0; t < 8; ++t) {
      float f0 = arow[kA + t];
      float f1 = arow[kA + 8 + t];
      __bf16 h0 = (__bf16)f0, h1 = (__bf16)f1;
      ah[t] = h0;            ah[8 + t] = h1;
      al[t] = (__bf16)(f0 - (float)h0);
      al[8 + t] = (__bf16)(f1 - (float)h1);
    }
    // ---- B tiles (32x16): lane = column; lane half selects K 0..15 / 16..31
    const int kB = kk + hi * 16;
    v16bf blh, bll, brh, brl;
#pragma unroll
    for (int t = 0; t < 16; ++t) {
      float fl = Wlin[(size_t)(kB + t) * DD + colB];
      float fr = Wres[(size_t)(kB + t) * DD + colB];
      __bf16 hl = (__bf16)fl, hr = (__bf16)fr;
      blh[t] = hl;  brh[t] = hr;
      bll[t] = (__bf16)(fl - (float)hl);
      brl[t] = (__bf16)(fr - (float)hr);
    }
    // ---- bf16x3: C += Ah*Bh + Al*Bh + Ah*Bl
    clin = __builtin_amdgcn_wmma_f32_16x16x32_bf16(false, ah, false, blh, (short)0, clin, false, false);
    clin = __builtin_amdgcn_wmma_f32_16x16x32_bf16(false, al, false, blh, (short)0, clin, false, false);
    clin = __builtin_amdgcn_wmma_f32_16x16x32_bf16(false, ah, false, bll, (short)0, clin, false, false);
    cres = __builtin_amdgcn_wmma_f32_16x16x32_bf16(false, ah, false, brh, (short)0, cres, false, false);
    cres = __builtin_amdgcn_wmma_f32_16x16x32_bf16(false, al, false, brh, (short)0, cres, false, false);
    cres = __builtin_amdgcn_wmma_f32_16x16x32_bf16(false, ah, false, brl, (short)0, cres, false, false);
  }

  // C/D layout: VGPR r -> row r (lanes 0-15) or row r+8 (lanes 16-31), col=lane%16
#pragma unroll
  for (int r = 0; r < 8; ++r) {
    const int row = row0 + r + hi * 8;
    x  [(size_t)row * DD + colB] = clin[r];
    res[(size_t)row * DD + colB] = cres[r];
  }
}

// ============================================================================
// Kernel 2: per-(node,head) attention logit halves
// ============================================================================
__global__ __launch_bounds__(256) void k_alpha(
    const float* __restrict__ x, const float* __restrict__ attl,
    const float* __restrict__ attr, float* __restrict__ al,
    float* __restrict__ ar) {
  const int id = blockIdx.x * blockDim.x + threadIdx.x;
  if (id >= NN * HH) return;
  const int h = id & (HH - 1);
  const int node = id >> 3;
  const float* xr = x + (size_t)node * DD + h * CC;
  float sl = 0.f, sr = 0.f;
#pragma unroll
  for (int c = 0; c < CC; ++c) {
    const float v = xr[c];
    sl += v * attl[h * CC + c];
    sr += v * attr[h * CC + c];
  }
  al[id] = sl;
  ar[id] = sr;
}

// ============================================================================
// Kernel 3: reset accumulators (must run every call; harness does not re-poison)
// ============================================================================
__global__ __launch_bounds__(256) void k_init(
    float* __restrict__ agg, int* __restrict__ nmax, float* __restrict__ nsum) {
  const int id = blockIdx.x * blockDim.x + threadIdx.x;
  if (id < NN * DD) agg[id] = 0.f;
  if (id < NN * HH) {
    nmax[id] = f2oi(__int_as_float(0xff800000u));  // -inf
    nsum[id] = 0.f;
  }
}

// ============================================================================
// Kernel 4: edge logits + per-(dst,head) running max
// ============================================================================
__global__ __launch_bounds__(256) void k_edge_logit(
    const int* __restrict__ ei, const float* __restrict__ ew,
    const float* __restrict__ al, const float* __restrict__ ar,
    float* __restrict__ logit, int* __restrict__ nmax) {
  const int id = blockIdx.x * blockDim.x + threadIdx.x;
  if (id >= EE * HH) return;
  const int e = id >> 3, h = id & 7;
  const int s = ei[e], d = ei[EE + e];
  float a = ew[e] * (al[s * HH + h] + ar[d * HH + h]);
  a = (a >= 0.f) ? a : 0.2f * a;  // leaky_relu(0.2)
  logit[id] = a;
  atomicMax(&nmax[d * HH + h], f2oi(a));
}

// ============================================================================
// Kernel 5: exp(logit - max) + per-(dst,head) sum
// ============================================================================
__global__ __launch_bounds__(256) void k_edge_exp(
    const int* __restrict__ ei, float* __restrict__ logit,
    const int* __restrict__ nmax, float* __restrict__ nsum) {
  const int id = blockIdx.x * blockDim.x + threadIdx.x;
  if (id >= EE * HH) return;
  const int e = id >> 3, h = id & 7;
  const int d = ei[EE + e];
  const float m = oi2f(nmax[d * HH + h]);
  const float ex = expf(logit[id] - m);
  logit[id] = ex;  // reuse buffer as exp values
  atomicAdd(&nsum[d * HH + h], ex);
}

// ============================================================================
// Kernel 6: message aggregation. One wave per edge, 4 floats per lane.
// ============================================================================
__global__ __launch_bounds__(256) void k_edge_agg(
    const int* __restrict__ ei, const float* __restrict__ ex,
    const float* __restrict__ nsum, const float* __restrict__ x,
    float* __restrict__ agg) {
  const int id = blockIdx.x * blockDim.x + threadIdx.x;
  const int e = id >> 5;
  if (e >= EE) return;
  const int lane = id & 31;
  const int q = lane * 4;   // 0..124
  const int h = q >> 4;     // head
  const int s = ei[e], d = ei[EE + e];
  const float coef = ex[e * HH + h] / nsum[d * HH + h];
  const float4 xv = *(const float4*)(x + (size_t)s * DD + q);
  float* ap = agg + (size_t)d * DD + q;
  atomicAdd(ap + 0, xv.x * coef);
  atomicAdd(ap + 1, xv.y * coef);
  atomicAdd(ap + 2, xv.z * coef);
  atomicAdd(ap + 3, xv.w * coef);
}

// ============================================================================
// Kernel 7: out = elu(agg) + res  (res was written into d_out by k_gemm_dual)
// ============================================================================
__global__ __launch_bounds__(256) void k_final(
    const float* __restrict__ agg, float* __restrict__ out) {
  const int id = blockIdx.x * blockDim.x + threadIdx.x;
  if (id >= NN * DD) return;
  const float a = agg[id];
  const float el = (a > 0.f) ? a : (expf(a) - 1.f);
  out[id] = el + out[id];
}

extern "C" void kernel_launch(void* const* d_in, const int* in_sizes, int n_in,
                              void* d_out, int out_size, void* d_ws, size_t ws_size,
                              hipStream_t stream) {
  const float* feat = (const float*)d_in[0];
  const int*   ei   = (const int*)d_in[1];
  const float* ew   = (const float*)d_in[2];
  const float* Wlin = (const float*)d_in[3];
  const float* attl = (const float*)d_in[4];
  const float* attr = (const float*)d_in[5];
  const float* Wres = (const float*)d_in[6];
  float* out = (float*)d_out;

  // workspace layout (floats): x[N*D] | logit/ex[E*H] | al[N*H] | ar[N*H] |
  //                            nmax[N*H] (as int) | nsum[N*H] | agg[N*D]
  float* ws    = (float*)d_ws;
  float* x     = ws;
  float* logit = x + (size_t)NN * DD;
  float* al    = logit + (size_t)EE * HH;
  float* ar    = al + (size_t)NN * HH;
  int*   nmax  = (int*)(ar + (size_t)NN * HH);
  float* nsum  = (float*)(nmax + (size_t)NN * HH);
  float* agg   = nsum + (size_t)NN * HH;

  k_gemm_dual<<<NN / 16, 256, 0, stream>>>(feat, Wlin, Wres, x, out);
  k_alpha<<<(NN * HH + 255) / 256, 256, 0, stream>>>(x, attl, attr, al, ar);
  k_init<<<(NN * DD + 255) / 256, 256, 0, stream>>>(agg, nmax, nsum);
  k_edge_logit<<<(EE * HH + 255) / 256, 256, 0, stream>>>(ei, ew, al, ar, logit, nmax);
  k_edge_exp<<<(EE * HH + 255) / 256, 256, 0, stream>>>(ei, logit, nmax, nsum);
  k_edge_agg<<<(EE * 32 + 255) / 256, 256, 0, stream>>>(ei, logit, nsum, x, agg);
  k_final<<<(NN * DD + 255) / 256, 256, 0, stream>>>(agg, out);
}
